// LSTMAttentionDot_15023795602102
// MI455X (gfx1250) — matrix operand order repaired and checked by reading, not compiled, not run on hardware
//
#include <hip/hip_runtime.h>
#include <hip/hip_bf16.h>
#include <cstdint>

// Problem constants
#define Bn 128
#define Tn 256
#define Sn 256
#define In 512
#define Hn 512
#define G4 2048   // 4*H

typedef __attribute__((ext_vector_type(16))) __bf16 v16bf;
typedef __attribute__((ext_vector_type(8)))  __bf16 v8bf;
typedef __attribute__((ext_vector_type(8)))  float  v8f;

// ---------------------------------------------------------------------------
// Converters / init
// ---------------------------------------------------------------------------
__global__ void k_f32_to_bf16(const float* __restrict__ src, __bf16* __restrict__ dst, int n) {
    int i = blockIdx.x * blockDim.x + threadIdx.x;
    if (i < n) dst[i] = (__bf16)src[i];
}

// ctx [S,B,H] (f32) -> ctx_b [B,S,H] (bf16)
__global__ void k_ctx_convert(const float* __restrict__ ctx, __bf16* __restrict__ dst) {
    int i = blockIdx.x * blockDim.x + threadIdx.x;   // over B*S*H
    if (i >= Bn * Sn * Hn) return;
    int h = i % Hn;
    int s = (i / Hn) % Sn;
    int b = i / (Hn * Sn);
    dst[i] = (__bf16)ctx[((size_t)s * Bn + b) * Hn + h];
}

__global__ void k_bias_sum(const float* __restrict__ bi, const float* __restrict__ bh,
                           float* __restrict__ bsum) {
    int i = blockIdx.x * blockDim.x + threadIdx.x;
    if (i < G4) bsum[i] = bi[i] + bh[i];
}

__global__ void k_init_state(const float* __restrict__ h0, const float* __restrict__ c0,
                             float* __restrict__ hf, __bf16* __restrict__ hb,
                             float* __restrict__ c) {
    int i = blockIdx.x * blockDim.x + threadIdx.x;
    if (i < Bn * Hn) {
        float h = h0[i];
        hf[i] = h;
        hb[i] = (__bf16)h;
        c[i]  = c0[i];
    }
}

// ---------------------------------------------------------------------------
// K1: fused gates GEMM (WMMA bf16) + LSTM cell + target GEMM (WMMA bf16)
//   gates[b,n] = sum_k x[b,t,k]*Wi[n,k] + sum_k h[b,k]*Wh[n,k]  (+bias in cell)
//   then hy staged in LDS and target = hy @ W_in^T for the same 16 rows.
//   grid = B/16 blocks, 256 threads (8 waves).
//   Gates pass: wave w owns columns {g*512 + w*64 + 0..63 : g=0..3} -> all 4
//   gates for hidden slice j in [w*64, w*64+64): cell runs on live accums.
//   Target pass: wave w owns columns [w*64, w*64+64).
// ---------------------------------------------------------------------------
__global__ __launch_bounds__(256)
void k_gates_cell_target(const __bf16* __restrict__ xb,    // [B,T,I] bf16
                         const __bf16* __restrict__ Wib,   // [4H,I]  bf16
                         const __bf16* __restrict__ Whb,   // [4H,H]  bf16
                         const __bf16* __restrict__ hcb,   // [B,H]   bf16 carry
                         const float*  __restrict__ bsum,  // [4H]
                         const __bf16* __restrict__ Winb,  // [H,H]   bf16
                         float*        __restrict__ cst,   // [B,H] c state (in/out)
                         __bf16*       __restrict__ concat,// [B,2H]: [:,H:] = hy
                         float*        __restrict__ targ,  // [B,H] target out
                         int t)
{
    __shared__ __bf16 s_hy[16 * Hn];     // 16KB: hy tile for this row block

    const int wv   = threadIdx.x >> 5;   // wave 0..7
    const int lane = threadIdx.x & 31;
    const int msel = lane >> 4;          // k-group selector for A/B frags
    const int mr   = lane & 15;          // row (A) / col (B) within tile
    const int m0   = blockIdx.x * 16;

    // ---------------- gates GEMM ----------------
    v8f acc[16] = {};                    // 16 accumulator tiles (i,f,g,o x 4)

    #pragma unroll
    for (int phase = 0; phase < 2; ++phase) {
        const __bf16* Arow = phase ? (hcb + (size_t)(m0 + mr) * Hn)
                                   : (xb + ((size_t)(m0 + mr) * Tn + t) * In);
        const __bf16* W    = phase ? Whb : Wib;

        for (int k0 = 0; k0 < 512; k0 += 32) {
            v16bf afrag;
            {
                const __bf16* pa = Arow + k0 + msel * 8;
                v8bf lo = *(const v8bf*)(pa);
                v8bf hi = *(const v8bf*)(pa + 16);
                #pragma unroll
                for (int e = 0; e < 8; ++e) { afrag[e] = lo[e]; afrag[8 + e] = hi[e]; }
            }
            #pragma unroll
            for (int nt = 0; nt < 16; ++nt) {
                const int gsel = nt >> 2;
                const int sub  = nt & 3;
                const int n    = gsel * 512 + wv * 64 + sub * 16 + mr;
                const __bf16* pb = W + (size_t)n * 512 + k0 + msel * 8;
                v8bf b0 = *(const v8bf*)(pb);
                v8bf b1 = *(const v8bf*)(pb + 16);
                v16bf bfrag;
                #pragma unroll
                for (int e = 0; e < 8; ++e) { bfrag[e] = b0[e]; bfrag[8 + e] = b1[e]; }
                acc[nt] = __builtin_amdgcn_wmma_f32_16x16x32_bf16(
                    false, afrag, false, bfrag, (short)0, acc[nt], false, false);
            }
        }
    }

    // ---------------- LSTM cell on live accumulators ----------------
    // C-layout: (vgpr v, lane l) = row m0 + v + 8*(l>=16), col base + (l&15)
    #pragma unroll
    for (int sub = 0; sub < 4; ++sub) {
        const int j = wv * 64 + sub * 16 + mr;           // hidden index 0..511
        #pragma unroll
        for (int v = 0; v < 8; ++v) {
            const int vrow = v + msel * 8;               // 0..15
            const int m    = m0 + vrow;                  // batch row
            float ig = acc[0  + sub][v] + bsum[j];
            float fg = acc[4  + sub][v] + bsum[512 + j];
            float gg = acc[8  + sub][v] + bsum[1024 + j];
            float og = acc[12 + sub][v] + bsum[1536 + j];
            float is = 1.0f / (1.0f + __expf(-ig));
            float fs = 1.0f / (1.0f + __expf(-fg));
            float os = 1.0f / (1.0f + __expf(-og));
            float gt = tanhf(gg);
            size_t idx = (size_t)m * Hn + j;
            float cy = fs * cst[idx] + is * gt;
            float hy = os * tanhf(cy);
            cst[idx] = cy;
            __bf16 hyb = (__bf16)hy;
            s_hy[vrow * Hn + j] = hyb;                   // LDS stage for target GEMM
            concat[(size_t)m * (2 * Hn) + Hn + j] = hyb; // [wc,hy] second half
        }
    }
    __syncthreads();

    // ---------------- target GEMM: target[16,512] = hy @ W_in^T ----------------
    v8f acc2[4] = {};
    for (int k0 = 0; k0 < 512; k0 += 32) {
        v16bf afrag;
        {
            const __bf16* pa = s_hy + mr * Hn + k0 + msel * 8;
            v8bf lo = *(const v8bf*)(pa);
            v8bf hi = *(const v8bf*)(pa + 16);
            #pragma unroll
            for (int e = 0; e < 8; ++e) { afrag[e] = lo[e]; afrag[8 + e] = hi[e]; }
        }
        #pragma unroll
        for (int nt = 0; nt < 4; ++nt) {
            const int n = wv * 64 + nt * 16 + mr;
            const __bf16* pb = Winb + (size_t)n * 512 + k0 + msel * 8;
            v8bf b0 = *(const v8bf*)(pb);
            v8bf b1 = *(const v8bf*)(pb + 16);
            v16bf bfrag;
            #pragma unroll
            for (int e = 0; e < 8; ++e) { bfrag[e] = b0[e]; bfrag[8 + e] = b1[e]; }
            acc2[nt] = __builtin_amdgcn_wmma_f32_16x16x32_bf16(
                false, afrag, false, bfrag, (short)0, acc2[nt], false, false);
        }
    }
    #pragma unroll
    for (int nt = 0; nt < 4; ++nt) {
        #pragma unroll
        for (int v = 0; v < 8; ++v) {
            const int m = m0 + v + msel * 8;
            const int n = wv * 64 + nt * 16 + mr;
            targ[(size_t)m * Hn + n] = acc2[nt][v];
        }
    }
}

// ---------------------------------------------------------------------------
// K2: per-batch attention core: scores = ctx_b . target ; softmax ;
//     wc = attn @ ctx_b  -> written as bf16 into concat[:, 0:H].
//     One WG per batch; ctx (bf16, 32MB) is L2-resident (192MB L2).
//     Scores are wave-cooperative with lane-contiguous ctx reads.
// ---------------------------------------------------------------------------
__global__ __launch_bounds__(256)
void k_attn(const float*  __restrict__ targ,   // [B,H]
            const __bf16* __restrict__ ctxb,   // [B,S,H]
            __bf16*       __restrict__ concat) // [B,2H]: [:,0:H] = wc
{
    __shared__ float s_tg[Hn];
    __shared__ float s_sc[Sn];
    __shared__ float s_r0[8];
    __shared__ float s_r1[8];

    const int b    = blockIdx.x;
    const int tid  = threadIdx.x;
    const int wv   = tid >> 5;
    const int lane = tid & 31;
    const __bf16* cb = ctxb + (size_t)b * Sn * Hn;

    for (int j = tid; j < Hn; j += 256) s_tg[j] = targ[(size_t)b * Hn + j];
    __syncthreads();

    // scores[s] = sum_h ctx[b,s,h]*target[h]; wave wv covers s in [wv*32, +32)
    for (int i = 0; i < 32; ++i) {
        const int s = wv * 32 + i;
        const __bf16* cr = cb + (size_t)s * Hn;
        float a = 0.f;
        for (int h = lane; h < Hn; h += 32) a += (float)cr[h] * s_tg[h];
        #pragma unroll
        for (int o = 16; o > 0; o >>= 1) a += __shfl_xor(a, o, 32);
        if (lane == 0) s_sc[s] = a;
    }
    __syncthreads();

    // softmax over S=256 (one value per thread)
    float v = s_sc[tid];
    float mx = v;
    #pragma unroll
    for (int o = 16; o > 0; o >>= 1) mx = fmaxf(mx, __shfl_xor(mx, o, 32));
    if (lane == 0) s_r0[wv] = mx;
    __syncthreads();
    if (tid == 0) {
        float m = s_r0[0];
        #pragma unroll
        for (int i = 1; i < 8; ++i) m = fmaxf(m, s_r0[i]);
        s_r0[0] = m;
    }
    __syncthreads();
    float e = __expf(v - s_r0[0]);
    float sm = e;
    #pragma unroll
    for (int o = 16; o > 0; o >>= 1) sm += __shfl_xor(sm, o, 32);
    if (lane == 0) s_r1[wv] = sm;
    __syncthreads();
    if (tid == 0) {
        float s = 0.f;
        #pragma unroll
        for (int i = 0; i < 8; ++i) s += s_r1[i];
        s_r1[0] = s;
    }
    __syncthreads();
    s_sc[tid] = e / s_r1[0];
    __syncthreads();

    // wc[j] = sum_s attn[s]*ctx[b,s,j]  (lane-contiguous j -> coalesced)
    for (int j = tid; j < Hn; j += 256) {
        float a = 0.f;
        for (int s = 0; s < Sn; ++s) a += s_sc[s] * (float)cb[(size_t)s * Hn + j];
        concat[(size_t)b * (2 * Hn) + j] = (__bf16)a;
    }
}

// ---------------------------------------------------------------------------
// K3: output projection GEMM (WMMA bf16) + fused tanh:
//   h_tilde = tanh([wc,hy] @ W_out^T)  : [128,1024] x [1024,512]
//   grid = B/16 blocks, 8 waves; wave owns 64 output cols (4 tiles), K=1024.
//   Writes output[b,t,:] and both h-carry buffers.
// ---------------------------------------------------------------------------
__global__ __launch_bounds__(256)
void k_out_proj(const __bf16* __restrict__ concat, // [B,2H] bf16
                const __bf16* __restrict__ Woutb,  // [H,2H] bf16
                float*        __restrict__ out,    // [B,T,H]
                float*        __restrict__ hcf,    // [B,H] carry f32
                __bf16*       __restrict__ hcb,    // [B,H] carry bf16
                int t)
{
    const int wv   = threadIdx.x >> 5;
    const int lane = threadIdx.x & 31;
    const int msel = lane >> 4;
    const int mr   = lane & 15;
    const int m0   = blockIdx.x * 16;

    v8f acc[4] = {};
    const __bf16* Arow = concat + (size_t)(m0 + mr) * (2 * Hn);

    for (int k0 = 0; k0 < 2 * Hn; k0 += 32) {
        v16bf afrag;
        {
            const __bf16* pa = Arow + k0 + msel * 8;
            v8bf lo = *(const v8bf*)(pa);
            v8bf hi = *(const v8bf*)(pa + 16);
            #pragma unroll
            for (int e = 0; e < 8; ++e) { afrag[e] = lo[e]; afrag[8 + e] = hi[e]; }
        }
        #pragma unroll
        for (int nt = 0; nt < 4; ++nt) {
            const int n = wv * 64 + nt * 16 + mr;
            const __bf16* pb = Woutb + (size_t)n * (2 * Hn) + k0 + msel * 8;
            v8bf b0 = *(const v8bf*)(pb);
            v8bf b1 = *(const v8bf*)(pb + 16);
            v16bf bfrag;
            #pragma unroll
            for (int e = 0; e < 8; ++e) { bfrag[e] = b0[e]; bfrag[8 + e] = b1[e]; }
            acc[nt] = __builtin_amdgcn_wmma_f32_16x16x32_bf16(
                false, afrag, false, bfrag, (short)0, acc[nt], false, false);
        }
    }

    #pragma unroll
    for (int nt = 0; nt < 4; ++nt) {
        #pragma unroll
        for (int v = 0; v < 8; ++v) {
            const int m = m0 + v + msel * 8;
            const int n = wv * 64 + nt * 16 + mr;
            float ht = tanhf(acc[nt][v]);
            out[((size_t)m * Tn + t) * Hn + n] = ht;
            size_t idx = (size_t)m * Hn + n;
            hcf[idx] = ht;
            hcb[idx] = (__bf16)ht;
        }
    }
}

// ---------------------------------------------------------------------------
// Host side
// ---------------------------------------------------------------------------
extern "C" void kernel_launch(void* const* d_in, const int* in_sizes, int n_in,
                              void* d_out, int out_size, void* d_ws, size_t ws_size,
                              hipStream_t stream) {
    (void)in_sizes; (void)n_in; (void)out_size; (void)ws_size;

    const float* x    = (const float*)d_in[0];   // [B,T,I]
    const float* h0   = (const float*)d_in[1];   // [B,H]
    const float* c0   = (const float*)d_in[2];   // [B,H]
    const float* ctx  = (const float*)d_in[3];   // [S,B,H]
    const float* Wi   = (const float*)d_in[4];   // [4H,I]
    const float* bi   = (const float*)d_in[5];   // [4H]
    const float* Wh   = (const float*)d_in[6];   // [4H,H]
    const float* bh   = (const float*)d_in[7];   // [4H]
    const float* W_in = (const float*)d_in[8];   // [H,H]
    const float* W_out= (const float*)d_in[9];   // [H,2H]
    float* out = (float*)d_out;

    // Workspace carve-out (256B aligned regions)
    uintptr_t base = (uintptr_t)d_ws;
    size_t off = 0;
    auto carve = [&](size_t bytes) -> uintptr_t {
        uintptr_t p = base + off;
        off += (bytes + 255) & ~(size_t)255;
        return p;
    };
    __bf16* xb    = (__bf16*)carve((size_t)Bn * Tn * In * 2);    // 33.5 MB
    __bf16* Wib   = (__bf16*)carve((size_t)G4 * In * 2);         //  2 MB
    __bf16* Whb   = (__bf16*)carve((size_t)G4 * Hn * 2);         //  2 MB
    __bf16* Winb  = (__bf16*)carve((size_t)Hn * Hn * 2);         //  0.5 MB
    __bf16* Woutb = (__bf16*)carve((size_t)Hn * 2 * Hn * 2);     //  1 MB
    __bf16* ctxb  = (__bf16*)carve((size_t)Bn * Sn * Hn * 2);    // 33.5 MB
    float*  bsum  = (float*) carve((size_t)G4 * 4);
    float*  hcf   = (float*) carve((size_t)Bn * Hn * 4);         // carry h f32
    __bf16* hcb   = (__bf16*)carve((size_t)Bn * Hn * 2);         // carry h bf16
    float*  cst   = (float*) carve((size_t)Bn * Hn * 4);         // c state
    float*  targ  = (float*) carve((size_t)Bn * Hn * 4);         // target
    __bf16* conc  = (__bf16*)carve((size_t)Bn * 2 * Hn * 2);     // [wc, hy] bf16

    // One-time conversions
    {
        int n = Bn * Tn * In;
        k_f32_to_bf16<<<(n + 255) / 256, 256, 0, stream>>>(x, xb, n);
    }
    {
        int n = G4 * In;
        k_f32_to_bf16<<<(n + 255) / 256, 256, 0, stream>>>(Wi, Wib, n);
    }
    {
        int n = G4 * Hn;
        k_f32_to_bf16<<<(n + 255) / 256, 256, 0, stream>>>(Wh, Whb, n);
    }
    {
        int n = Hn * Hn;
        k_f32_to_bf16<<<(n + 255) / 256, 256, 0, stream>>>(W_in, Winb, n);
    }
    {
        int n = Hn * 2 * Hn;
        k_f32_to_bf16<<<(n + 255) / 256, 256, 0, stream>>>(W_out, Woutb, n);
    }
    {
        int n = Bn * Sn * Hn;
        k_ctx_convert<<<(n + 255) / 256, 256, 0, stream>>>(ctx, ctxb);
    }
    k_bias_sum<<<(G4 + 255) / 256, 256, 0, stream>>>(bi, bh, bsum);
    k_init_state<<<(Bn * Hn + 255) / 256, 256, 0, stream>>>(h0, c0, hcf, hcb, cst);

    // Sequential scan: 3 fused kernels per timestep (all GEMMs on WMMA)
    for (int t = 0; t < Tn; ++t) {
        k_gates_cell_target<<<Bn / 16, 256, 0, stream>>>(xb, Wib, Whb, hcb, bsum,
                                                         Winb, cst, conc, targ, t);
        k_attn<<<Bn, 256, 0, stream>>>(targ, ctxb, conc);
        k_out_proj<<<Bn / 16, 256, 0, stream>>>(conc, Woutb, out, hcf, hcb, t);
    }

    // Final carries: hf then cf appended after output[B,T,H]
    hipMemcpyAsync(out + (size_t)Bn * Tn * Hn, hcf,
                   (size_t)Bn * Hn * sizeof(float), hipMemcpyDeviceToDevice, stream);
    hipMemcpyAsync(out + (size_t)Bn * Tn * Hn + (size_t)Bn * Hn, cst,
                   (size_t)Bn * Hn * sizeof(float), hipMemcpyDeviceToDevice, stream);
}
